// Encoder_5248450035714
// MI455X (gfx1250) — matrix operand order repaired and checked by reading, not compiled, not run on hardware
//
#include <hip/hip_runtime.h>
#include <hip/hip_bf16.h>

// ---------------------------------------------------------------------------
// 2-layer LSTM (PyTorch gate order i,f,g,o) for MI455X / gfx1250, wave32 WMMA.
// S=256, B=64, E=H=1024. bf16 GEMM inputs, f32 accumulate (v_wmma_f32_16x16x32_bf16).
// Recurrent step stages W_hh tiles in LDS via GLOBAL_LOAD_ASYNC_TO_LDS_B128.
// ---------------------------------------------------------------------------

#define S_LEN 256
#define B_DIM 64
#define E_DIM 1024
#define H_DIM 1024
#define G_DIM 4096  // 4*H

typedef __bf16 bf16_t;
typedef __attribute__((ext_vector_type(16))) __bf16 bf16x16;
typedef __attribute__((ext_vector_type(8)))  __bf16 bf16x8;
typedef __attribute__((ext_vector_type(8)))  float  floatx8;

#define HAS_ASYNC_LDS __has_builtin(__builtin_amdgcn_global_load_async_to_lds_b128)
// Builtin parameter types (per hipcc diagnostic): v4i in AS1 / AS3.
typedef int v4i_t __attribute__((__vector_size__(16)));
typedef __attribute__((address_space(1))) v4i_t* gas_b128;
typedef __attribute__((address_space(3))) v4i_t* las_b128;

// LDS row pitch for staged W_hh rows: 1024 + 8 bf16 -> 516 dwords/row, bank
// offset 4 per row => half-wave ds_load_b128 touches all 64 banks.
#define ROWPAD 1032
#define STEP_LDS_BYTES (8 * 16 * ROWPAD * 2)  // 8 groups x 16 rows, 264192 B

// ---- WMMA fragment loaders (contiguous k order shared by A and B, wave32) --------------------

// A: 16x32 bf16 tile, rows m0..m0+15, cols k0..k0+31 of row-major [*, ld].
// lane<16: M=lane, elems 0..15 = K k0..k0+15
// lane>=16: M=lane-16, elems 0..15 = K k0+16..k0+31
__device__ __forceinline__ bf16x16 load_frag_a(const bf16_t* __restrict__ base,
                                               int ld, int m0, int k0, int lane) {
    int m  = m0 + (lane & 15);
    int kh = (lane >> 4) << 4;  // 0 or 16
    const bf16_t* p = base + (size_t)m * ld + k0 + kh;
    union { bf16x16 v; bf16x8 h[2]; } u;
    u.h[0] = *(const bf16x8*)(p);        // 16B aligned
    u.h[1] = *(const bf16x8*)(p + 8);   // 16B aligned
    return u.v;
}

// B: 32x16 tile where B[k][n] = W[n][k], W row-major [N, ld].
// lane<16: N=n0+lane,  K = k0..k0+15  (one contiguous 32B load)
// lane>=16: N=n0+lane-16, K = k0+16..k0+31
__device__ __forceinline__ bf16x16 load_frag_b(const bf16_t* __restrict__ W,
                                               int ld, int n0, int k0, int lane) {
    int n    = n0 + (lane & 15);
    int koff = k0 + ((lane & 16) ? 16 : 0);
    return *(const bf16x16*)(W + (size_t)n * ld + koff);  // 32B aligned
}

// Same B layout but from an LDS-staged group (16 rows, ROWPAD pitch).
__device__ __forceinline__ bf16x16 load_frag_b_lds(const bf16_t* grp,
                                                   int k0, int lane) {
    int n    = lane & 15;
    int koff = k0 + ((lane & 16) ? 16 : 0);
    const bf16_t* p = grp + (size_t)n * ROWPAD + koff;
    union { bf16x16 v; bf16x8 h[2]; } u;
    u.h[0] = *(const bf16x8*)(p);       // ds_load_b128
    u.h[1] = *(const bf16x8*)(p + 8);   // ds_load_b128
    return u.v;
}

__device__ __forceinline__ floatx8 wmma_bf16(bf16x16 a, bf16x16 b, floatx8 c) {
    return __builtin_amdgcn_wmma_f32_16x16x32_bf16(
        /*neg_a=*/false, a, /*neg_b=*/false, b,
        /*c_mod=*/(short)0, c, /*reuse_a=*/false, /*reuse_b=*/false);
}

__device__ __forceinline__ float sigmoidf_fast(float x) {
    return 1.0f / (1.0f + __expf(-x));
}

// ---- Utility kernels -------------------------------------------------------

__global__ void convert_f32_bf16_kernel(const float* __restrict__ src,
                                        bf16_t* __restrict__ dst, int n) {
    int i = blockIdx.x * blockDim.x + threadIdx.x;
    if (i < n) dst[i] = (bf16_t)src[i];
}

__global__ void zero_bf16_kernel(bf16_t* __restrict__ p, int n) {
    int i = blockIdx.x * blockDim.x + threadIdx.x;
    if (i < n) p[i] = (bf16_t)0.0f;
}

__global__ void zero_f32_kernel(float* __restrict__ p, int n) {
    int i = blockIdx.x * blockDim.x + threadIdx.x;
    if (i < n) p[i] = 0.0f;
}

// Embedding gather + f32 -> bf16. One block per (s,b) row.
__global__ void gather_embed_kernel(const int* __restrict__ seq,
                                    const float* __restrict__ emb,
                                    bf16_t* __restrict__ X) {
    int row = blockIdx.x;                 // 0 .. S*B-1
    int tok = seq[row];
    const float* src = emb + (size_t)tok * E_DIM;
    bf16_t* dst = X + (size_t)row * E_DIM;
    for (int i = threadIdx.x; i < E_DIM; i += blockDim.x)
        dst[i] = (bf16_t)src[i];
}

// ---- x_proj GEMM: C[M,4096] = X[M,1024](bf16) * W[4096,1024]^T + b_ih + b_hh
// 256 threads = 8 waves. Wave computes a 64(M) x 16(N) strip (4 WMMAs / k-step).
// Grid: (N/128, M/64) = (32, 256).
__global__ void __launch_bounds__(256) xproj_gemm_kernel(
    const bf16_t* __restrict__ X,
    const bf16_t* __restrict__ W,
    const float* __restrict__ b_ih, const float* __restrict__ b_hh,
    float* __restrict__ C) {
    const int K = 1024;
    int wave = threadIdx.x >> 5;
    int lane = threadIdx.x & 31;
    int n0 = (blockIdx.x * 8 + wave) * 16;
    int m0 = blockIdx.y * 64;

    floatx8 acc0 = {}, acc1 = {}, acc2 = {}, acc3 = {};
    for (int k0 = 0; k0 < K; k0 += 32) {
        bf16x16 bfr = load_frag_b(W, K, n0, k0, lane);
        bf16x16 a0  = load_frag_a(X, K, m0 +  0, k0, lane);
        bf16x16 a1  = load_frag_a(X, K, m0 + 16, k0, lane);
        bf16x16 a2  = load_frag_a(X, K, m0 + 32, k0, lane);
        bf16x16 a3  = load_frag_a(X, K, m0 + 48, k0, lane);
        acc0 = wmma_bf16(a0, bfr, acc0);
        acc1 = wmma_bf16(a1, bfr, acc1);
        acc2 = wmma_bf16(a2, bfr, acc2);
        acc3 = wmma_bf16(a3, bfr, acc3);
    }

    int col   = n0 + (lane & 15);
    int rbase = m0 + ((lane >> 4) << 3);
    float bias = b_ih[col] + b_hh[col];
    float* Cc = C + col;
    floatx8 accs[4] = {acc0, acc1, acc2, acc3};
    for (int mt = 0; mt < 4; ++mt)
        for (int j = 0; j < 8; ++j)
            Cc[(size_t)(rbase + mt * 16 + j) * G_DIM] = accs[mt][j] + bias;
}

// ---- LSTM recurrent step: gates = x_proj[t] + h_prev @ W_hh^T, then update.
// Block = 256 thr = 8 waves, covers 2 j-tiles x all 4 m-tiles; the block's
// W_hh slice (8 groups = 2 j-tiles x 4 gates, 16 rows x 1024 K each) is staged
// into LDS once via async global->LDS copies (ASYNCcnt), then the k-loop reads
// B fragments from LDS (removes the 4x redundant L2 reads across m-tiles).
// Wave owns (16 batch rows) x (16 hidden cols) and ALL FOUR gate tiles, so the
// elementwise i/f/g/o update is lane-local. Grid: H/32 = 32 blocks.
__global__ void __launch_bounds__(256) lstm_step_kernel(
    const bf16_t* __restrict__ h_prev,   // [64,1024] bf16
    const bf16_t* __restrict__ Whh,      // [4096,1024] bf16
    const float* __restrict__ xproj_t,   // [64,4096] f32 (this timestep)
    float* __restrict__ c,               // [64,1024] f32, in-place carry
    float* __restrict__ h_f32,           // [64,1024] f32, or nullptr (last step only)
    bf16_t* __restrict__ h_next,         // [64,1024] bf16 (double buffer)
    bf16_t* __restrict__ ys_t) {         // [64,1024] bf16 slot, or nullptr
    extern __shared__ char smem_raw[];
    bf16_t* sh = (bf16_t*)smem_raw;      // [8 groups][16 rows][ROWPAD]

    int j0blk = blockIdx.x * 32;         // block covers hidden cols j0blk..+31

    // ---- Stage W_hh rows for all 8 (jhalf,gate) groups: 16384 x 16B chunks.
    // group g: jhalf = g>>2, gate = g&3; rows n = gate*H + j0blk + jhalf*16 + r
    for (int chunk = threadIdx.x; chunk < 8 * 16 * 128; chunk += 256) {
        int g    = chunk >> 11;          // /2048
        int rem  = chunk & 2047;
        int row  = rem >> 7;             // /128
        int kc   = (rem & 127) << 3;     // element offset (8 bf16 = 16B)
        int gate = g & 3;
        int jh   = g >> 2;
        const bf16_t* src =
            Whh + (size_t)(gate * H_DIM + j0blk + jh * 16 + row) * H_DIM + kc;
        bf16_t* dst = sh + (size_t)(g * 16 + row) * ROWPAD + kc;
#if HAS_ASYNC_LDS
        __builtin_amdgcn_global_load_async_to_lds_b128(
            (gas_b128)src, (las_b128)dst, /*offset=*/0, /*cpol=*/0);
#else
        *(bf16x8*)dst = *(const bf16x8*)src;   // fallback: load + ds_store
#endif
    }
#if HAS_ASYNC_LDS
#if __has_builtin(__builtin_amdgcn_s_wait_asynccnt)
    __builtin_amdgcn_s_wait_asynccnt(0);
#else
    asm volatile("s_wait_asynccnt 0x0" ::: "memory");
#endif
#endif
    __syncthreads();

    int wave = threadIdx.x >> 5;
    int lane = threadIdx.x & 31;
    int m0    = (wave & 3) * 16;
    int jhalf = wave >> 2;
    const bf16_t* shj = sh + (size_t)(jhalf * 4 * 16) * ROWPAD;  // 4 gate groups

    floatx8 gi = {}, gf = {}, gg = {}, go = {};
    for (int k0 = 0; k0 < H_DIM; k0 += 32) {
        bf16x16 a  = load_frag_a(h_prev, H_DIM, m0, k0, lane);
        bf16x16 bi = load_frag_b_lds(shj + (size_t)0 * 16 * ROWPAD, k0, lane);
        bf16x16 bf = load_frag_b_lds(shj + (size_t)1 * 16 * ROWPAD, k0, lane);
        bf16x16 bg = load_frag_b_lds(shj + (size_t)2 * 16 * ROWPAD, k0, lane);
        bf16x16 bo = load_frag_b_lds(shj + (size_t)3 * 16 * ROWPAD, k0, lane);
        gi = wmma_bf16(a, bi, gi);
        gf = wmma_bf16(a, bf, gf);
        gg = wmma_bf16(a, bg, gg);
        go = wmma_bf16(a, bo, go);
    }

    int j     = j0blk + jhalf * 16 + (lane & 15);
    int rbase = m0 + ((lane >> 4) << 3);
    for (int r = 0; r < 8; ++r) {
        int b = rbase + r;
        const float* xp = xproj_t + (size_t)b * G_DIM;
        float vi = sigmoidf_fast(gi[r] + xp[0 * H_DIM + j]);
        float vf = sigmoidf_fast(gf[r] + xp[1 * H_DIM + j]);
        float vg = tanhf(gg[r] + xp[2 * H_DIM + j]);
        float vo = sigmoidf_fast(go[r] + xp[3 * H_DIM + j]);
        size_t idx = (size_t)b * H_DIM + j;
        float cv = vf * c[idx] + vi * vg;
        c[idx] = cv;
        float hv = vo * tanhf(cv);
        bf16_t hb = (bf16_t)hv;
        h_next[idx] = hb;
        if (ys_t)  ys_t[idx]  = hb;
        if (h_f32) h_f32[idx] = hv;
    }
}

// ---------------------------------------------------------------------------

extern "C" void kernel_launch(void* const* d_in, const int* in_sizes, int n_in,
                              void* d_out, int out_size, void* d_ws, size_t ws_size,
                              hipStream_t stream) {
    const int*   seq    = (const int*)d_in[0];
    const float* emb    = (const float*)d_in[1];
    const float* w_ih_0 = (const float*)d_in[2];
    const float* w_hh_0 = (const float*)d_in[3];
    const float* b_ih_0 = (const float*)d_in[4];
    const float* b_hh_0 = (const float*)d_in[5];
    const float* w_ih_1 = (const float*)d_in[6];
    const float* w_hh_1 = (const float*)d_in[7];
    const float* b_ih_1 = (const float*)d_in[8];
    const float* b_hh_1 = (const float*)d_in[9];

    // Output layout: h_n[2,64,1024] then c_n[2,64,1024] (flat f32).
    float* out    = (float*)d_out;
    float* h_out0 = out;                       // layer0 final h
    float* h_out1 = out + 64 * 1024;           // layer1 final h
    float* c_buf0 = out + 2 * 64 * 1024;       // layer0 c (carried in-place)
    float* c_buf1 = out + 3 * 64 * 1024;       // layer1 c (carried in-place)

    // Workspace carve-out (256B aligned). Total ~ 371 MB.
    char* ws = (char*)d_ws;
    size_t off = 0;
    auto carve = [&](size_t bytes) -> void* {
        void* p = ws + off;
        off += (bytes + 255) & ~(size_t)255;
        return p;
    };
    bf16_t* Xbf   = (bf16_t*)carve((size_t)S_LEN * B_DIM * E_DIM * 2); // 33.5 MB
    bf16_t* ysbf  = (bf16_t*)carve((size_t)S_LEN * B_DIM * H_DIM * 2); // 33.5 MB
    float*  xproj = (float*) carve((size_t)S_LEN * B_DIM * G_DIM * 4); // 268 MB
    bf16_t* wih0b = (bf16_t*)carve((size_t)G_DIM * E_DIM * 2);         // 8.4 MB
    bf16_t* whh0b = (bf16_t*)carve((size_t)G_DIM * H_DIM * 2);
    bf16_t* wih1b = (bf16_t*)carve((size_t)G_DIM * H_DIM * 2);
    bf16_t* whh1b = (bf16_t*)carve((size_t)G_DIM * H_DIM * 2);
    bf16_t* hA    = (bf16_t*)carve((size_t)B_DIM * H_DIM * 2);
    bf16_t* hB    = (bf16_t*)carve((size_t)B_DIM * H_DIM * 2);

    const int wn = G_DIM * 1024;  // elements per weight matrix
    convert_f32_bf16_kernel<<<wn / 256, 256, 0, stream>>>(w_ih_0, wih0b, wn);
    convert_f32_bf16_kernel<<<wn / 256, 256, 0, stream>>>(w_hh_0, whh0b, wn);
    convert_f32_bf16_kernel<<<wn / 256, 256, 0, stream>>>(w_ih_1, wih1b, wn);
    convert_f32_bf16_kernel<<<wn / 256, 256, 0, stream>>>(w_hh_1, whh1b, wn);

    gather_embed_kernel<<<S_LEN * B_DIM, 256, 0, stream>>>(seq, emb, Xbf);

    const int hn = B_DIM * H_DIM;  // 65536

    // ---- Layer 0 ----
    xproj_gemm_kernel<<<dim3(G_DIM / 128, (S_LEN * B_DIM) / 64), 256, 0, stream>>>(
        Xbf, wih0b, b_ih_0, b_hh_0, xproj);
    zero_bf16_kernel<<<hn / 256, 256, 0, stream>>>(hA, hn);
    zero_f32_kernel<<<hn / 256, 256, 0, stream>>>(c_buf0, hn);
    for (int t = 0; t < S_LEN; ++t) {
        bf16_t* hin  = (t & 1) ? hB : hA;
        bf16_t* hout = (t & 1) ? hA : hB;
        lstm_step_kernel<<<H_DIM / 32, 256, STEP_LDS_BYTES, stream>>>(
            hin, whh0b, xproj + (size_t)t * B_DIM * G_DIM,
            c_buf0, (t == S_LEN - 1) ? h_out0 : nullptr, hout,
            ysbf + (size_t)t * B_DIM * H_DIM);
    }

    // ---- Layer 1 (input = ys of layer 0, already bf16) ----
    xproj_gemm_kernel<<<dim3(G_DIM / 128, (S_LEN * B_DIM) / 64), 256, 0, stream>>>(
        ysbf, wih1b, b_ih_1, b_hh_1, xproj);
    zero_bf16_kernel<<<hn / 256, 256, 0, stream>>>(hA, hn);
    zero_f32_kernel<<<hn / 256, 256, 0, stream>>>(c_buf1, hn);
    for (int t = 0; t < S_LEN; ++t) {
        bf16_t* hin  = (t & 1) ? hB : hA;
        bf16_t* hout = (t & 1) ? hA : hB;
        lstm_step_kernel<<<H_DIM / 32, 256, STEP_LDS_BYTES, stream>>>(
            hin, whh1b, xproj + (size_t)t * B_DIM * G_DIM,
            c_buf1, (t == S_LEN - 1) ? h_out1 : nullptr, hout, nullptr);
    }
}
